// HybridEstimatorQNN_65481071408580
// MI455X (gfx1250) — compile-verified, matrix operand order checked by reading, not run on hardware
//
#include <hip/hip_runtime.h>
#include <hip/hip_bf16.h>
#include <cmath>

typedef __attribute__((ext_vector_type(16))) _Float16 v16h;
typedef __attribute__((ext_vector_type(8)))  _Float16 v8h;
typedef __attribute__((ext_vector_type(8)))  float    v8f;

// ---------------------------------------------------------------------------
// Kernel 1: precompute the 4x81 Pauli-polynomial coefficients C from theta.
//   U = fixed layer unitary (16x16 complex), A_w = Re(U^H D_w U),
//   C_w[t] = (1/16) * sum_{j,k} A_w[j,k] * prod_i sigma_{t_i}[j_i,k_i]
// ---------------------------------------------------------------------------
__global__ void __launch_bounds__(512) qnn_precompute(const float* __restrict__ theta,
                                                      float* __restrict__ Cg) {
  __shared__ float Ur[16][16], Ui[16][16];   // [basis s][column j]
  __shared__ float Am[4][16][16];
  const int tid = threadIdx.x;

  if (tid < 16) {
    float ar[16], ai[16];
    for (int s = 0; s < 16; ++s) { ar[s] = 0.f; ai[s] = 0.f; }
    ar[tid] = 1.f;

    auto ap1 = [&](int w, float u00r, float u00i, float u01r, float u01i,
                          float u10r, float u10i, float u11r, float u11i) {
      int st = 8 >> w;
      for (int i = 0; i < 16; ++i) if (!(i & st)) {
        int j = i | st;
        float a0r = ar[i], a0i = ai[i], a1r = ar[j], a1i = ai[j];
        ar[i] = u00r*a0r - u00i*a0i + u01r*a1r - u01i*a1i;
        ai[i] = u00r*a0i + u00i*a0r + u01r*a1i + u01i*a1r;
        ar[j] = u10r*a0r - u10i*a0i + u11r*a1r - u11i*a1i;
        ai[j] = u10r*a0i + u10i*a0r + u11r*a1i + u11i*a1r;
      }
    };
    auto cnot = [&](int c, int t) {
      int sc = 8 >> c, st = 8 >> t;
      for (int i = 0; i < 16; ++i) if ((i & sc) && !(i & st)) {
        int j = i | st;
        float tr = ar[i], ti = ai[i];
        ar[i] = ar[j]; ai[i] = ai[j]; ar[j] = tr; ai[j] = ti;
      }
    };

    float t0 = theta[0], t1 = theta[1], t2 = theta[2], t3 = theta[3], t4 = theta[4];
    { float c = cosf(0.5f*t0), s = sinf(0.5f*t0); ap1(0, c,0.f, -s,0.f,  s,0.f,  c,0.f); } // RY w0
    { float c = cosf(0.5f*t1), s = sinf(0.5f*t1); ap1(1, c,0.f, 0.f,-s, 0.f,-s,  c,0.f); } // RX w1
    cnot(0, 1);
    { float c = cosf(0.5f*t2), s = sinf(0.5f*t2); ap1(2, c,-s, 0.f,0.f, 0.f,0.f, c, s);  } // RZ w2
    cnot(2, 3);
    { float c = cosf(0.5f*t3), s = sinf(0.5f*t3); ap1(3, c,0.f, -s,0.f,  s,0.f,  c,0.f); } // RY w3
    cnot(1, 2);
    { float c = cosf(0.5f*t4), s = sinf(0.5f*t4); ap1(0, c,0.f, 0.f,-s, 0.f,-s,  c,0.f); } // RX w0

    for (int s = 0; s < 16; ++s) { Ur[s][tid] = ar[s]; Ui[s][tid] = ai[s]; }
  }
  __syncthreads();

  for (int e = tid; e < 4*16*16; e += blockDim.x) {
    int w = e >> 8, j = (e >> 4) & 15, k = e & 15;
    float acc = 0.f;
    for (int s = 0; s < 16; ++s) {
      float sign = ((s >> (3 - w)) & 1) ? -1.f : 1.f;
      acc += sign * (Ur[s][j]*Ur[s][k] + Ui[s][j]*Ui[s][k]);
    }
    Am[w][j][k] = acc;
  }
  __syncthreads();

  for (int e = tid; e < 4*81; e += blockDim.x) {
    int w = e / 81, t = e % 81;
    int tt[4]; { int tmp = t; tt[3] = tmp%3; tmp/=3; tt[2] = tmp%3; tmp/=3; tt[1] = tmp%3; tt[0] = tmp/3; }
    float acc = 0.f;
    for (int j = 0; j < 16; ++j)
      for (int k = 0; k < 16; ++k) {
        float pr = Am[w][j][k];
        for (int i = 0; i < 4; ++i) {
          int ji = (j >> (3 - i)) & 1, ki = (k >> (3 - i)) & 1, ti = tt[i];
          float sv;
          if (ti == 0)      sv = (ji == ki) ? 1.f : 0.f;                 // I
          else if (ti == 1) sv = (ji == ki) ? (ji ? -1.f : 1.f) : 0.f;   // Z -> cos
          else              sv = (ji != ki) ? 1.f : 0.f;                 // X -> sin
          pr *= sv;
        }
        acc += pr;
      }
    Cg[w*81 + t] = acc * (1.f / 16.f);
  }
}

// ---------------------------------------------------------------------------
// Kernel 2: convert W1 (256x784 -> 256x800 zero-padded f16) and W2 (64x256 f16)
// ---------------------------------------------------------------------------
__global__ void qnn_convert(const float* __restrict__ W1, const float* __restrict__ W2,
                            _Float16* __restrict__ W1h, _Float16* __restrict__ W2h) {
  int i = blockIdx.x * blockDim.x + threadIdx.x;
  const int n1 = 256 * 800;
  if (i < n1) {
    int o = i / 800, k = i % 800;
    W1h[i] = (k < 784) ? (_Float16)W1[o*784 + k] : (_Float16)0.f;
  } else if (i < n1 + 64*256) {
    int j = i - n1;
    W2h[j] = (_Float16)W2[j];
  }
}

// ---------------------------------------------------------------------------
// Kernel 3: quanv as 81-term polynomial; writes f16 features (B x 800, padded)
// ---------------------------------------------------------------------------
__global__ void __launch_bounds__(256) qnn_quanv(const float* __restrict__ x,
                                                 const float* __restrict__ Cg,
                                                 _Float16* __restrict__ fh, int B) {
  __shared__ float sC[324];
  for (int i = threadIdx.x; i < 324; i += blockDim.x) sC[i] = Cg[i];
  __syncthreads();

  int i = blockIdx.x * blockDim.x + threadIdx.x;
  if (i >= B * 200) return;
  int b = i / 200, q = i % 200;
  _Float16* dst = fh + (size_t)b * 800 + q * 4;
  if (q >= 196) {  // zero pad columns 784..799
    dst[0] = (_Float16)0.f; dst[1] = (_Float16)0.f;
    dst[2] = (_Float16)0.f; dst[3] = (_Float16)0.f;
    return;
  }
  int r = q / 14, c = q % 14;
  const float* xb = x + (size_t)b * 784;
  float p0 = xb[(2*r)  *28 + 2*c    ];
  float p1 = xb[(2*r)  *28 + 2*c + 1];
  float p2 = xb[(2*r+1)*28 + 2*c    ];
  float p3 = xb[(2*r+1)*28 + 2*c + 1];
  float c0, s0, c1, s1, c2, s2, c3, s3;
  sincosf(p0, &s0, &c0); sincosf(p1, &s1, &c1);
  sincosf(p2, &s2, &c2); sincosf(p3, &s3, &c3);

  float ab[9] = {1.f, c1, s1, c0, c0*c1, c0*s1, s0, s0*c1, s0*s1};
  float cd[9] = {1.f, c3, s3, c2, c2*c3, c2*s3, s2, s2*c3, s2*s3};

  float e0 = 0.f, e1 = 0.f, e2 = 0.f, e3 = 0.f;
#pragma unroll
  for (int u = 0; u < 9; ++u) {
#pragma unroll
    for (int v = 0; v < 9; ++v) {
      float pr = ab[u] * cd[v];
      int t = u * 9 + v;
      e0 = fmaf(sC[t],       pr, e0);
      e1 = fmaf(sC[81 + t],  pr, e1);
      e2 = fmaf(sC[162 + t], pr, e2);
      e3 = fmaf(sC[243 + t], pr, e3);
    }
  }
  dst[0] = (_Float16)e0; dst[1] = (_Float16)e1;
  dst[2] = (_Float16)e2; dst[3] = (_Float16)e3;
}

// ---------------------------------------------------------------------------
// CDNA5 async copy: global -> LDS, 16B per active lane, tracked by ASYNCcnt
// (CDNA5 ISA). LDS byte address = low 32 bits of the
// generic shared pointer (aperture scheme).
// ---------------------------------------------------------------------------
__device__ inline void async_load_b128(const void* gaddr, void* ldsaddr) {
  unsigned loff = (unsigned)(unsigned long long)ldsaddr;
  asm volatile("global_load_async_to_lds_b128 %0, %1, off"
               :: "v"(loff), "v"(gaddr) : "memory");
}
__device__ inline void wait_async0() {
  asm volatile("s_wait_asynccnt 0" ::: "memory");
}

// ---------------------------------------------------------------------------
// WMMA helpers (layouts per CDNA5 ISA)
// ---------------------------------------------------------------------------
__device__ inline v16h frag_from_lds(const _Float16* base) {
  v8h lo = *(const v8h*)base;
  v8h hi = *(const v8h*)(base + 16);
  return __builtin_shufflevector(lo, hi, 0,1,2,3,4,5,6,7,8,9,10,11,12,13,14,15);
}

#define LDS_STRIDE 40  // 32 + 8 halves: 80B rows -> conflict-free b128 LDS reads

// ---------------------------------------------------------------------------
// Kernel 4: GEMM1  h1 = relu(f @ W1^T + b1) -> f16   M=B, N=256, K=800
// 256 threads (8 waves), 64x64 tile, 2 WMMA tiles/wave, double-buffered
// async global->LDS staging, one barrier per K-step.
// ---------------------------------------------------------------------------
__global__ void __launch_bounds__(256) qnn_gemm1(const _Float16* __restrict__ fh,
                                                 const _Float16* __restrict__ W1h,
                                                 const float* __restrict__ b1,
                                                 _Float16* __restrict__ h1h, int B) {
  __shared__ _Float16 sA[2][64 * LDS_STRIDE];
  __shared__ _Float16 sB[2][64 * LDS_STRIDE];
  const int tid  = threadIdx.x;
  const int lane = tid & 31, wave = tid >> 5;
  const int row  = tid >> 2, quad = tid & 3;
  const int ln = lane & 15, lh = lane >> 4;
  const int mt0 = wave >> 2, nt = wave & 3;       // tiles (mt0,nt) and (mt0+2,nt)
  const int mblk = blockIdx.x * 64, nblk = blockIdx.y * 64;

  int gm = mblk + row; if (gm >= B) gm = B - 1;   // clamp: OOB rows never stored
  const _Float16* gA = fh  + (size_t)gm * 800 + quad * 8;
  const _Float16* gB = W1h + (size_t)(nblk + row) * 800 + quad * 8;
  _Float16* dA0 = &sA[0][row * LDS_STRIDE + quad * 8];
  _Float16* dA1 = &sA[1][row * LDS_STRIDE + quad * 8];
  _Float16* dB0 = &sB[0][row * LDS_STRIDE + quad * 8];
  _Float16* dB1 = &sB[1][row * LDS_STRIDE + quad * 8];

  v8f acc0 = {}; v8f acc1 = {};

  async_load_b128(gA, dA0);
  async_load_b128(gB, dB0);

  for (int kt = 0; kt < 25; ++kt) {
    const int cur = kt & 1;
    wait_async0();        // my tile-(kt) chunks have landed in LDS
    __syncthreads();      // everyone's landed; everyone done reading buf cur^1
    if (kt + 1 < 25) {
      async_load_b128(gA + (kt + 1) * 32, cur ? dA0 : dA1);
      async_load_b128(gB + (kt + 1) * 32, cur ? dB0 : dB1);
    }
    v16h a0 = frag_from_lds(&sA[cur][(mt0*16 + ln) * LDS_STRIDE + 8*lh]);
    v16h a1 = frag_from_lds(&sA[cur][((mt0+2)*16 + ln) * LDS_STRIDE + 8*lh]);
    v16h bb = frag_from_lds(&sB[cur][(nt*16 + ln) * LDS_STRIDE + 8*lh]);
    acc0 = __builtin_amdgcn_wmma_f32_16x16x32_f16(false, a0, false, bb, (short)0, acc0, false, false);
    acc1 = __builtin_amdgcn_wmma_f32_16x16x32_f16(false, a1, false, bb, (short)0, acc1, false, false);
  }

  const int gn = nblk + nt*16 + ln;
  const float bias = b1[gn];
  const int mb0 = mblk + mt0*16 + 8*lh;
  const int mb1 = mblk + (mt0+2)*16 + 8*lh;
#pragma unroll
  for (int r = 0; r < 8; ++r) {
    int g0 = mb0 + r;
    if (g0 < B) h1h[(size_t)g0 * 256 + gn] = (_Float16)fmaxf(acc0[r] + bias, 0.f);
    int g1 = mb1 + r;
    if (g1 < B) h1h[(size_t)g1 * 256 + gn] = (_Float16)fmaxf(acc1[r] + bias, 0.f);
  }
}

// ---------------------------------------------------------------------------
// Kernel 5: GEMM2 (h2 = relu(h1 @ W2^T + b2)) fused with GEMM3 (out = h2@W3^T+b3)
// M=B, N=64 (full width per block), K=256; same async double-buffered core.
// ---------------------------------------------------------------------------
__global__ void __launch_bounds__(256) qnn_gemm2(const _Float16* __restrict__ h1h,
                                                 const _Float16* __restrict__ W2h,
                                                 const float* __restrict__ b2,
                                                 const float* __restrict__ W3,
                                                 const float* __restrict__ b3,
                                                 float* __restrict__ out, int B) {
  __shared__ _Float16 sA[2][64 * LDS_STRIDE];
  __shared__ _Float16 sB[2][64 * LDS_STRIDE];
  __shared__ float sH[64 * 68];
  const int tid  = threadIdx.x;
  const int lane = tid & 31, wave = tid >> 5;
  const int row  = tid >> 2, quad = tid & 3;
  const int ln = lane & 15, lh = lane >> 4;
  const int mt0 = wave >> 2, nt = wave & 3;
  const int mblk = blockIdx.x * 64;

  int gm = mblk + row; if (gm >= B) gm = B - 1;
  const _Float16* gA = h1h + (size_t)gm * 256 + quad * 8;
  const _Float16* gB = W2h + (size_t)row * 256 + quad * 8;
  _Float16* dA0 = &sA[0][row * LDS_STRIDE + quad * 8];
  _Float16* dA1 = &sA[1][row * LDS_STRIDE + quad * 8];
  _Float16* dB0 = &sB[0][row * LDS_STRIDE + quad * 8];
  _Float16* dB1 = &sB[1][row * LDS_STRIDE + quad * 8];

  v8f acc0 = {}; v8f acc1 = {};

  async_load_b128(gA, dA0);
  async_load_b128(gB, dB0);

  for (int kt = 0; kt < 8; ++kt) {
    const int cur = kt & 1;
    wait_async0();
    __syncthreads();
    if (kt + 1 < 8) {
      async_load_b128(gA + (kt + 1) * 32, cur ? dA0 : dA1);
      async_load_b128(gB + (kt + 1) * 32, cur ? dB0 : dB1);
    }
    v16h a0 = frag_from_lds(&sA[cur][(mt0*16 + ln) * LDS_STRIDE + 8*lh]);
    v16h a1 = frag_from_lds(&sA[cur][((mt0+2)*16 + ln) * LDS_STRIDE + 8*lh]);
    v16h bb = frag_from_lds(&sB[cur][(nt*16 + ln) * LDS_STRIDE + 8*lh]);
    acc0 = __builtin_amdgcn_wmma_f32_16x16x32_f16(false, a0, false, bb, (short)0, acc0, false, false);
    acc1 = __builtin_amdgcn_wmma_f32_16x16x32_f16(false, a1, false, bb, (short)0, acc1, false, false);
  }

  const int gn = nt*16 + ln;          // local column 0..63
  const float bias = b2[gn];
  const int ml0 = mt0*16 + 8*lh;
  const int ml1 = (mt0+2)*16 + 8*lh;
  __syncthreads();                    // sA/sB dead; reuse LDS phase for sH
#pragma unroll
  for (int r = 0; r < 8; ++r) {
    sH[(ml0 + r) * 68 + gn] = fmaxf(acc0[r] + bias, 0.f);
    sH[(ml1 + r) * 68 + gn] = fmaxf(acc1[r] + bias, 0.f);
  }
  __syncthreads();

  if (tid < 64) {
    float s = b3[0];
#pragma unroll
    for (int j = 0; j < 64; ++j) s = fmaf(sH[tid * 68 + j], W3[j], s);
    int g = mblk + tid;
    if (g < B) out[g] = s;
  }
}

// ---------------------------------------------------------------------------
extern "C" void kernel_launch(void* const* d_in, const int* in_sizes, int n_in,
                              void* d_out, int out_size, void* d_ws, size_t ws_size,
                              hipStream_t stream) {
  const float* x     = (const float*)d_in[0];
  const float* theta = (const float*)d_in[1];
  const float* W1    = (const float*)d_in[2];
  const float* b1    = (const float*)d_in[3];
  const float* W2    = (const float*)d_in[4];
  const float* b2    = (const float*)d_in[5];
  const float* W3    = (const float*)d_in[6];
  const float* b3    = (const float*)d_in[7];
  float* out = (float*)d_out;

  const int B = in_sizes[0] / 784;

  char* ws = (char*)d_ws;
  float*     Cg  = (float*)(ws + 0);                 // 4*81 f32
  _Float16*  W1h = (_Float16*)(ws + 2048);           // 256*800 f16
  _Float16*  W2h = (_Float16*)(ws + 411648);         // 64*256 f16
  _Float16*  fh  = (_Float16*)(ws + 444416);         // B*800 f16
  _Float16*  h1h = (_Float16*)(ws + 444416 + (size_t)B * 800 * 2);  // B*256 f16

  qnn_precompute<<<1, 512, 0, stream>>>(theta, Cg);

  const int ncv = 256 * 800 + 64 * 256;
  qnn_convert<<<(ncv + 255) / 256, 256, 0, stream>>>(W1, W2, W1h, W2h);

  qnn_quanv<<<(B * 200 + 255) / 256, 256, 0, stream>>>(x, Cg, fh, B);

  qnn_gemm1<<<dim3((B + 63) / 64, 4), 256, 0, stream>>>(fh, W1h, b1, h1h, B);

  qnn_gemm2<<<(B + 63) / 64, 256, 0, stream>>>(h1h, W2h, b2, W3, b3, out, B);
}
